// VisualSemanticMemoryScorer_29850022707717
// MI455X (gfx1250) — matrix operand order repaired and checked by reading, not compiled, not run on hardware
//
#include <hip/hip_runtime.h>

// ---------------------------------------------------------------------------
// VisualSemanticMemoryScorer on MI455X (gfx1250), wave32, BF16 WMMA.
//  * All large time-parallel matmuls -> k_gemm using v_wmma_f32_16x16x32_bf16,
//    vectorized b128 fragment loads, uniform-guard-free inner loops.
//  * Prototype/semantic algebra folded into tiny per-row mixes (k_fold/k_semin).
//  * Serial GRU/slow-memory scans: both paths in one kernel (grid=2), 16 waves,
//    per-step WMMA with M=16 (4 live rows), weights hot in L2, states in LDS,
//    global_prefetch of next step's gi/xs rows to hide L2 latency.
//  * x-dependent projections (gi, xs) hoisted out of the serial loop.
// ---------------------------------------------------------------------------

#define D_ 2048
#define F_ 256
#define H_ 256
#define L_ 384
#define C_ 48
#define S_ 19
#define E_ 7
#define A_ 8
#define B_ 4
#define T_ 2048
#define M_TOT (B_*T_)          // 8192 rows (b*T + t)
#define CAP_ 0.2f
#define TAU_STEP_ 0.07f
#define TAU_ERR_ 0.07f
#define PROTO_TAU_ 0.07f

typedef __attribute__((ext_vector_type(8)))  float          v8f;
typedef __attribute__((ext_vector_type(16))) __bf16         v16bf;
typedef __attribute__((ext_vector_type(16))) unsigned short v16u;
typedef __attribute__((ext_vector_type(8)))  unsigned short v8u;
typedef unsigned short bfu;

__device__ inline bfu f2bf(float f){
  unsigned u = __float_as_uint(f);
  u += 0x7FFFu + ((u >> 16) & 1u);          // round-to-nearest-even
  return (bfu)(u >> 16);
}
__device__ inline float bf2f(bfu h){ return __uint_as_float(((unsigned)h) << 16); }
__device__ inline float geluf(float x){ return 0.5f*x*(1.0f + erff(x*0.70710678118f)); }
__device__ inline float sigm(float x){ return 1.0f/(1.0f + __expf(-x)); }

// --------------------------- WMMA fragment loads ---------------------------
// A (16x32 bf16, row-major src, ld multiple of 32): lane = half*16 + row.
// Lane data = two contiguous 16B chunks: K in [half*16, half*16+8) and +8.
__device__ inline v16bf load_a_frag(const bfu* src, int ld, int k0, int lane){
  int row  = lane & 15;
  int half = lane >> 4;
  const bfu* p = src + (size_t)row*ld + k0 + half*16;
  v8u lo = *(const v8u*)(p);         // -> global_load_b128
  v8u hi = *(const v8u*)(p + 8);     // -> global_load_b128
  v16u au = __builtin_shufflevector(lo, hi, 0,1,2,3,4,5,6,7,8,9,10,11,12,13,14,15);
  return __builtin_bit_cast(v16bf, au);
}
// B (32x16 bf16) from W[N,K] row-major (B = W^T): col n = n0 + (lane&15),
// lanes 0-15 carry K=k0..k0+15, lanes 16-31 carry K=k0+16..k0+31. One 32B load.
__device__ inline v16bf load_b_frag(const bfu* w, int ld, int n0, int k0, int lane){
  const bfu* p = w + (size_t)(n0 + (lane & 15))*ld + k0 + (lane >> 4)*16;
  v16u bu = *(const v16u*)p;
  return __builtin_bit_cast(v16bf, bu);
}

// --------------------------- batch GEMM kernel -----------------------------
// Out[M,N] (f32, ld=N) (+)= A[M,K]bf16 @ W[N,K]bf16^T (+bias).
// block = 256 thr = 8 waves, wave handles two 16-wide N tiles; grid=(M/16, ceil(N/256)).
template<int NT>
__device__ inline void gemm_core(const bfu* __restrict__ Arow, const bfu* __restrict__ W,
                                 int K, int n0, int n1, int lane, v8f* acc0, v8f* acc1){
  for (int k0 = 0; k0 < K; k0 += 32){
    v16bf a = load_a_frag(Arow, K, k0, lane);
    {
      v16bf b = load_b_frag(W, K, n0, k0, lane);
      *acc0 = __builtin_amdgcn_wmma_f32_16x16x32_bf16(false, a, false, b, (short)0, *acc0, false, false);
    }
    if (NT == 2){
      v16bf b = load_b_frag(W, K, n1, k0, lane);
      *acc1 = __builtin_amdgcn_wmma_f32_16x16x32_bf16(false, a, false, b, (short)0, *acc1, false, false);
    }
  }
}
__global__ void k_gemm(const bfu* __restrict__ A, const bfu* __restrict__ W,
                       const float* __restrict__ bias, float* __restrict__ Out,
                       int M, int N, int K, int accum){
  int lane = threadIdx.x & 31, wave = threadIdx.x >> 5;
  int m0  = blockIdx.x * 16;
  int nt0 = blockIdx.y * 16 + wave*2;
  int n0 = nt0*16, n1 = n0 + 16;
  bool do0 = n0 < N, do1 = n1 < N;
  if (!do0) return;
  v8f acc0 = {}; v8f acc1 = {};
  const bfu* Arow = A + (size_t)m0 * K;
  if (do1) gemm_core<2>(Arow, W, K, n0, n1, lane, &acc0, &acc1);
  else     gemm_core<1>(Arow, W, K, n0, n1, lane, &acc0, &acc1);
  int col = lane & 15, half = lane >> 4;
  #pragma unroll
  for (int r = 0; r < 8; ++r){
    int row = m0 + half*8 + r;
    {
      int c = n0 + col; float v = acc0[r] + (bias ? bias[c] : 0.f);
      float* o = Out + (size_t)row*N + c; *o = accum ? (*o + v) : v;
    }
    if (do1){
      int c = n1 + col; float v = acc1[r] + (bias ? bias[c] : 0.f);
      float* o = Out + (size_t)row*N + c; *o = accum ? (*o + v) : v;
    }
  }
}

// --------------------------- elementwise / prep ----------------------------
__global__ void k_f2bf(const float* __restrict__ s, bfu* __restrict__ d, size_t n){
  size_t i = (size_t)blockIdx.x*blockDim.x + threadIdx.x;
  if (i < n) d[i] = f2bf(s[i]);
}
__global__ void k_zero_bf(bfu* p, size_t n){
  size_t i = (size_t)blockIdx.x*blockDim.x + threadIdx.x;
  if (i < n) p[i] = 0;
}
// slice cols [c0,c0+Kd) of src[N,ldK] -> dst[N,Kd] bf16
__global__ void k_f2bf_slice(const float* __restrict__ s, bfu* __restrict__ d,
                             int N, int ldK, int c0, int Kd){
  int i = blockIdx.x*blockDim.x + threadIdx.x;
  if (i < N*Kd){
    int n = i / Kd, k = i % Kd;
    d[i] = f2bf(s[(size_t)n*ldK + c0 + k]);
  }
}
// features [B,D,T] -> bf16 [B*T, D]
__global__ void k_feat_tr(const float* __restrict__ f, bfu* __restrict__ o){
  size_t i = (size_t)blockIdx.x*blockDim.x + threadIdx.x;
  if (i < (size_t)B_*T_*D_){
    size_t r = i / D_; int d = (int)(i % D_);
    int b = (int)(r / T_), t = (int)(r % T_);
    o[i] = f2bf(f[((size_t)b*D_ + d)*T_ + t]);
  }
}
// per-(b,t): ||x_t|| and x_t . x_{t-1} (consistency feature), on f32 input
__global__ void k_rowstats(const float* __restrict__ f, float* norm, float* dotp){
  int r = blockIdx.x; int b = r / T_, t = r % T_;
  const float* base = f + (size_t)b*D_*T_ + t;
  float s = 0.f, dp = 0.f;
  for (int d = threadIdx.x; d < D_; d += blockDim.x){
    float x = base[(size_t)d*T_];
    s += x*x;
    if (t > 0) dp += x * base[(size_t)d*T_ - 1];
  }
  __shared__ float sh[256], sh2[256];
  sh[threadIdx.x] = s; sh2[threadIdx.x] = dp; __syncthreads();
  for (int o2 = 128; o2 > 0; o2 >>= 1){
    if (threadIdx.x < o2){ sh[threadIdx.x] += sh[threadIdx.x+o2]; sh2[threadIdx.x] += sh2[threadIdx.x+o2]; }
    __syncthreads();
  }
  if (threadIdx.x == 0){ norm[r] = sqrtf(fmaxf(sh[0], 1e-16f)); dotp[r] = sh2[0]; }
}
// L2-normalize rows of f32 src -> bf16 dst (rows < rows; dst pre-zeroed for pads)
__global__ void k_norm_rows(const float* __restrict__ src, bfu* __restrict__ dst, int rows, int K){
  int r = blockIdx.x; if (r >= rows) return;
  __shared__ float sh[256];
  float s = 0.f;
  for (int k = threadIdx.x; k < K; k += 256){ float v = src[(size_t)r*K + k]; s += v*v; }
  sh[threadIdx.x] = s; __syncthreads();
  for (int o = 128; o > 0; o >>= 1){ if (threadIdx.x < o) sh[threadIdx.x] += sh[threadIdx.x+o]; __syncthreads(); }
  float inv = rsqrtf(fmaxf(sh[0], 1e-16f));
  for (int k = threadIdx.x; k < K; k += 256) dst[(size_t)r*K + k] = f2bf(src[(size_t)r*K + k]*inv);
}
// fold prototype mixing into sin_W:  G[s<19] = P_s@W[:, :2048]^T + W[:,4104+s]^T ;
// G[19+e] = P_e@W[:,2048:4096]^T
__global__ void k_fold(const float* __restrict__ Ps, const float* __restrict__ Pe,
                       const float* __restrict__ sinW, float* __restrict__ G){
  int i = blockIdx.x*blockDim.x + threadIdx.x;
  if (i >= (S_+E_)*F_) return;
  int s = i / F_, f = i % F_;
  const float* wrow = sinW + (size_t)f*(2*D_ + A_ + S_);
  float acc;
  if (s < S_){
    acc = wrow[2*D_ + A_ + s];
    const float* p = Ps + (size_t)s*D_;
    for (int d = 0; d < D_; ++d) acc += p[d]*wrow[d];
  } else {
    acc = 0.f;
    const float* p = Pe + (size_t)(s - S_)*D_;
    for (int d = 0; d < D_; ++d) acc += p[d]*wrow[D_ + d];
  }
  G[i] = acc;
}
// softmaxes + aux stats + folded sem_in (block per row, 256 thr)
__global__ void k_semin(const float* __restrict__ sims, const float* __restrict__ norm,
                        const float* __restrict__ dotp, const float* __restrict__ G,
                        const float* __restrict__ sinW, const float* __restrict__ sin_b,
                        float* __restrict__ auxout, bfu* __restrict__ semin){
  int r = blockIdx.x; int t = r % T_;
  __shared__ float spost[S_], epost[E_], aux[A_];
  if (threadIdx.x == 0){
    float inv = 1.0f / fmaxf(norm[r], 1e-8f);
    float cs[S_], ce[E_];
    float mx = -1e30f, mean_s = 0.f;
    for (int s = 0; s < S_; ++s){ float c = sims[(size_t)r*32 + s]*inv; cs[s] = c/TAU_STEP_; mean_s += c; mx = fmaxf(mx, cs[s]); }
    mean_s /= (float)S_;
    float Z = 0.f;
    for (int s = 0; s < S_; ++s){ float e = __expf(cs[s]-mx); cs[s] = e; Z += e; }
    float t1 = -1.f, t2 = -1.f, ent = 0.f, prog = 0.f;
    for (int s = 0; s < S_; ++s){
      float p = cs[s]/Z; spost[s] = p;
      ent -= p*__logf(p + 1e-8f); prog += p*(float)s;
      if (p > t1){ t2 = t1; t1 = p; } else if (p > t2) t2 = p;
    }
    prog /= (float)(S_-1);
    float mxe = -1e30f;
    for (int e = 0; e < E_; ++e){ float c = sims[(size_t)r*32 + S_ + e]*inv; ce[e] = c/TAU_ERR_; mxe = fmaxf(mxe, ce[e]); }
    float Ze = 0.f;
    for (int e = 0; e < E_; ++e){ float x = __expf(ce[e]-mxe); ce[e] = x; Ze += x; }
    float ente = 0.f, emax = 0.f;
    for (int e = 0; e < E_; ++e){
      float p = ce[e]/Ze; epost[e] = p;
      ente -= p*__logf(p + 1e-8f); emax = fmaxf(emax, p);
    }
    float cons = (t == 0) ? 1.0f : dotp[r]*inv*(1.0f/fmaxf(norm[r-1], 1e-8f));
    aux[0]=t1; aux[1]=ent; aux[2]=emax; aux[3]=ente; aux[4]=mean_s; aux[5]=t1-t2; aux[6]=prog; aux[7]=cons;
    for (int a = 0; a < A_; ++a) auxout[(size_t)r*A_ + a] = aux[a];
  }
  __syncthreads();
  int f = threadIdx.x;
  float acc = sin_b[f];
  for (int s = 0; s < S_; ++s) acc += spost[s]*G[s*F_ + f];
  for (int e = 0; e < E_; ++e) acc += epost[e]*G[(S_+e)*F_ + f];
  const float* wrow = sinW + (size_t)f*(2*D_ + A_ + S_) + 2*D_;
  for (int a = 0; a < A_; ++a) acc += aux[a]*wrow[a];
  semin[(size_t)r*F_ + f] = f2bf(geluf(acc));
}
// LayerNorm (+ optional residual) + activation; block per row.
// act: 0=none 1=gelu ; outputs optional f32 and/or bf16.
__global__ void k_ln_act(const float* __restrict__ X, const float* __restrict__ Rres,
                         const float* __restrict__ g, const float* __restrict__ be,
                         int N, int act, float* __restrict__ outF, bfu* __restrict__ outB){
  int r = blockIdx.x;
  const float* x = X + (size_t)r*N;
  const float* rr = Rres ? Rres + (size_t)r*N : nullptr;
  __shared__ float sh[256], sh2[256];
  float s = 0.f, s2 = 0.f;
  for (int j = threadIdx.x; j < N; j += 256){
    float v = x[j] + (rr ? rr[j] : 0.f);
    s += v; s2 += v*v;
  }
  sh[threadIdx.x] = s; sh2[threadIdx.x] = s2; __syncthreads();
  for (int o = 128; o > 0; o >>= 1){
    if (threadIdx.x < o){ sh[threadIdx.x] += sh[threadIdx.x+o]; sh2[threadIdx.x] += sh2[threadIdx.x+o]; }
    __syncthreads();
  }
  float mean = sh[0]/(float)N;
  float var  = sh2[0]/(float)N - mean*mean;
  float inv  = rsqrtf(var + 1e-5f);
  for (int j = threadIdx.x; j < N; j += 256){
    float v = x[j] + (rr ? rr[j] : 0.f);
    v = (v - mean)*inv*g[j] + be[j];
    if (act == 1) v = geluf(v);
    if (outF) outF[(size_t)r*N + j] = v;
    if (outB) outB[(size_t)r*N + j] = f2bf(v);
  }
}
// small K (=8) dense: out[r,f] (+)= sum_a x[r,a]*W[f*ldW+c0+a] (+bias)
__global__ void k_lin_small(const float* __restrict__ x, int xld,
                            const float* __restrict__ W, int wld, int c0,
                            const float* __restrict__ bias, float* __restrict__ out, int accum){
  int r = blockIdx.x, f = threadIdx.x;
  float acc = bias ? bias[f] : 0.f;
  #pragma unroll
  for (int a = 0; a < A_; ++a) acc += x[(size_t)r*xld + a]*W[(size_t)f*wld + c0 + a];
  float* o = out + (size_t)r*F_ + f;
  *o = accum ? (*o + acc) : acc;
}
__global__ void k_gatemul(const float* __restrict__ gpre, const float* __restrict__ sctx,
                          bfu* __restrict__ out, size_t n){
  size_t i = (size_t)blockIdx.x*blockDim.x + threadIdx.x;
  if (i < n) out[i] = f2bf(sigm(gpre[i])*sctx[i]);
}
__global__ void k_rownorm_cvt(const float* __restrict__ src, bfu* __restrict__ dst,
                              float* __restrict__ norms, int N){
  int r = blockIdx.x;
  __shared__ float sh[256];
  float s = 0.f;
  for (int j = threadIdx.x; j < N; j += 256){
    float v = src[(size_t)r*N + j]; s += v*v;
    dst[(size_t)r*N + j] = f2bf(v);
  }
  sh[threadIdx.x] = s; __syncthreads();
  for (int o = 128; o > 0; o >>= 1){ if (threadIdx.x < o) sh[threadIdx.x] += sh[threadIdx.x+o]; __syncthreads(); }
  if (threadIdx.x == 0) norms[r] = sqrtf(fmaxf(sh[0], 1e-16f));
}
__global__ void k_finalize(const float* __restrict__ mainl, const float* __restrict__ pg2,
                           const float* __restrict__ pg_b2, const float* __restrict__ qdot,
                           const float* __restrict__ qnorm, const int* __restrict__ ids,
                           float* __restrict__ out){
  size_t i = (size_t)blockIdx.x*blockDim.x + threadIdx.x;
  if (i >= (size_t)B_*C_*T_) return;
  int t = (int)(i % T_); size_t bc = i / T_;
  int c = (int)(bc % C_), b = (int)(bc / C_);
  size_t r = (size_t)b*T_ + t;
  float v = mainl[r*C_ + c];
  float invq = 1.0f/(fmaxf(qnorm[r], 1e-8f)*PROTO_TAU_);
  for (int s = 0; s < S_; ++s){
    if (ids[s] == c)
      v += sigm(pg2[r*32 + s] + pg_b2[s]) * (qdot[r*32 + s]*invq);
  }
  out[i] = v;
}
__global__ void k_frames(const float* __restrict__ jointF, float* __restrict__ out2){
  size_t i = (size_t)blockIdx.x*blockDim.x + threadIdx.x;
  if (i >= (size_t)B_*F_*T_) return;
  int t = (int)(i % T_); size_t bf = i / T_;
  int f = (int)(bf % F_), b = (int)(bf / F_);
  out2[i] = jointF[((size_t)b*T_ + t)*F_ + f];
}

// ------------------------------- the scans ---------------------------------
struct ScanArgs {
  const float* gi;   // [M,768] = x@Wih^T + bih (precomputed)
  const float* xs;   // [M,384] = x@W1x^T + b1 (precomputed)
  const bfu*   Whh;  const float* bhh;      // [768,256]
  const bfu*   W1h;                          // [384,256]
  const float* sum_g; const float* sum_be;   // LN(384)
  const bfu*   W2;   const float* b2;        // [384,384]
  const bfu*   Wg_s; const bfu* Wg_l; const float* bg;  // [384,384] each
  const bfu*   Wc_s; const bfu* Wc_l; const float* bc;
  bfu*         hl;   // [M,640] out: h | l (bf16, feeds ctx GEMM)
};
// per-wave WMMA GEMM over LDS A (16 rows, only 0..3 live) x global W[N,K]
__device__ inline void scan_gemm(const bfu* Alds, int lda, const bfu* W, int K, int N,
                                 float* outLds, int accum, int wave, int lane){
  int ntiles = N/16;
  for (int nt = wave; nt < ntiles; nt += 16){
    v8f acc = {};
    for (int k0 = 0; k0 < K; k0 += 32){
      v16bf a = load_a_frag(Alds, lda, k0, lane);
      v16bf b = load_b_frag(W, K, nt*16, k0, lane);
      acc = __builtin_amdgcn_wmma_f32_16x16x32_bf16(false, a, false, b, (short)0, acc, false, false);
    }
    if (lane < 16){                 // half==0 -> rows 0..7 in acc[0..7]; rows 0..3 live
      #pragma unroll
      for (int r = 0; r < 4; ++r){
        float* o = outLds + r*N + nt*16 + lane;
        *o = accum ? (*o + acc[r]) : acc[r];
      }
    }
  }
}
__global__ __launch_bounds__(512)
void k_scan(ScanArgs va, ScanArgs sa){
  ScanArgs P = (blockIdx.x == 0) ? va : sa;
  const int tid = threadIdx.x, nthr = 512;
  const int lane = tid & 31, wave = tid >> 5;
  __shared__ bfu   hA[16*H_];      // 16x256 bf16 A staging (rows 4..15 zero)
  __shared__ bfu   sA[16*L_];      // 16x384
  __shared__ bfu   lA[16*L_];      // 16x384
  __shared__ float h32[B_*H_], l32[B_*L_];
  __shared__ float gh[B_*3*H_];
  __shared__ float sb[B_*L_], gb[B_*L_], cb[B_*L_];
  __shared__ float mstat[B_*2];
  for (int i = tid; i < 16*H_; i += nthr) hA[i] = 0;
  for (int i = tid; i < 16*L_; i += nthr){ sA[i] = 0; lA[i] = 0; }
  for (int i = tid; i < B_*H_; i += nthr) h32[i] = 0.f;
  for (int i = tid; i < B_*L_; i += nthr) l32[i] = 0.f;
  __syncthreads();

  for (int t = 0; t < T_; ++t){
    // prefetch next step's streamed activations (gi: 4x768 f32, xs: 4x384 f32)
    // one global_prefetch_b8 per 128B cacheline, hidden under this step's WMMAs
    if (t + 1 < T_){
      if (tid < B_*24){            // gi: 768*4B = 24 lines per batch row
        int b = tid / 24, ln = tid % 24;
        __builtin_prefetch(P.gi + ((size_t)b*T_ + t + 1)*(3*H_) + ln*32, 0, 1);
      } else if (tid < B_*24 + B_*12){   // xs: 384*4B = 12 lines per batch row
        int q = tid - B_*24; int b = q / 12, ln = q % 12;
        __builtin_prefetch(P.xs + ((size_t)b*T_ + t + 1)*L_ + ln*32, 0, 1);
      }
    }
    // gh = h @ Whh^T
    scan_gemm(hA, H_, P.Whh, H_, 3*H_, gh, 0, wave, lane);
    __syncthreads();
    // GRU gates
    for (int idx = tid; idx < B_*H_; idx += nthr){
      int b = idx / H_, j = idx % H_;
      const float* gir = P.gi + ((size_t)b*T_ + t)*(3*H_);
      float hr = gh[b*3*H_ + j]        + P.bhh[j];
      float hz = gh[b*3*H_ + H_  + j]  + P.bhh[H_ + j];
      float hn = gh[b*3*H_ + 2*H_ + j] + P.bhh[2*H_ + j];
      float rg = sigm(gir[j] + hr);
      float z  = sigm(gir[H_ + j] + hz);
      float n  = tanhf(gir[2*H_ + j] + rg*hn);
      float hnew = (1.f - z)*n + z*h32[idx];
      h32[idx] = hnew;
      hA[b*H_ + j] = f2bf(hnew);
    }
    __syncthreads();
    // summary pre = xs_t + h @ W1h^T ; LN + gelu -> sA
    scan_gemm(hA, H_, P.W1h, H_, L_, sb, 0, wave, lane);
    __syncthreads();
    if (wave < B_){
      int b = wave;
      float s = 0.f, s2 = 0.f;
      for (int j = lane; j < L_; j += 32){
        float v = sb[b*L_ + j] + P.xs[((size_t)b*T_ + t)*L_ + j];
        sb[b*L_ + j] = v; s += v; s2 += v*v;
      }
      for (int o = 16; o > 0; o >>= 1){ s += __shfl_down(s, o, 32); s2 += __shfl_down(s2, o, 32); }
      if (lane == 0){ mstat[b*2] = s/(float)L_; mstat[b*2+1] = s2/(float)L_; }
    }
    __syncthreads();
    for (int idx = tid; idx < B_*L_; idx += nthr){
      int b = idx / L_, j = idx % L_;
      float mean = mstat[b*2], var = mstat[b*2+1] - mean*mean;
      float v = (sb[idx] - mean)*rsqrtf(var + 1e-5f)*P.sum_g[j] + P.sum_be[j];
      sA[idx] = f2bf(geluf(v));
    }
    __syncthreads();
    // s2 = gelu(ln(...)) @ W2^T + b2  -> back into sA (bf16)
    scan_gemm(sA, L_, P.W2, L_, L_, sb, 0, wave, lane);
    __syncthreads();
    for (int idx = tid; idx < B_*L_; idx += nthr)
      sA[idx] = f2bf(sb[idx] + P.b2[idx % L_]);
    __syncthreads();
    // capped slow update: gate/cand = [s2,l] @ {Wg,Wc}
    scan_gemm(sA, L_, P.Wg_s, L_, L_, gb, 0, wave, lane);
    scan_gemm(lA, L_, P.Wg_l, L_, L_, gb, 1, wave, lane);
    scan_gemm(sA, L_, P.Wc_s, L_, L_, cb, 0, wave, lane);
    scan_gemm(lA, L_, P.Wc_l, L_, L_, cb, 1, wave, lane);
    __syncthreads();
    for (int idx = tid; idx < B_*L_; idx += nthr){
      int b = idx / L_, j = idx % L_;
      float gate = sigm(gb[idx] + P.bg[j])*CAP_;
      float cand = tanhf(cb[idx] + P.bc[j]);
      float lnew = (1.f - gate)*l32[idx] + gate*cand;
      l32[idx] = lnew;
      bfu lb = f2bf(lnew);
      lA[b*L_ + j] = lb;
      P.hl[((size_t)b*T_ + t)*640 + 256 + j] = lb;
    }
    for (int idx = tid; idx < B_*H_; idx += nthr){
      int b = idx / H_, j = idx % H_;
      P.hl[((size_t)b*T_ + t)*640 + j] = hA[b*H_ + j];
    }
    __syncthreads();
  }
}

// ------------------------------- host side ---------------------------------
extern "C" void kernel_launch(void* const* d_in, const int* in_sizes, int n_in,
                              void* d_out, int out_size, void* d_ws, size_t ws_size,
                              hipStream_t stream){
  (void)in_sizes; (void)n_in; (void)out_size; (void)ws_size;
  const float* feats = (const float*)d_in[0];
  const float* stepP = (const float*)d_in[1];
  const float* errP  = (const float*)d_in[2];
  const int*   ids   = (const int*)d_in[3];
  #define PRM(i) ((const float*)d_in[4 + (i)])
  enum {
    iBASE_W=0,iBASE_B,iBASE_G,iBASE_BE,
    iVGRU_WIH,iVGRU_BIH,iVGRU_WHH,iVGRU_BHH,
    iVSUM_W1,iVSUM_B1,iVSUM_G,iVSUM_BE,iVSUM_W2,iVSUM_B2,
    iVLONG_WG,iVLONG_BG,iVLONG_WC,iVLONG_BC,
    iVCTX_W,iVCTX_B,iVCTX_G,iVCTX_BE,
    iSIN_W,iSIN_B,
    iSGRU_WIH,iSGRU_BIH,iSGRU_WHH,iSGRU_BHH,
    iSSUM_W1,iSSUM_B1,iSSUM_G,iSSUM_BE,iSSUM_W2,iSSUM_B2,
    iSLONG_WG,iSLONG_BG,iSLONG_WC,iSLONG_BC,
    iSCTX_W,iSCTX_B,iSCTX_G,iSCTX_BE,
    iAUX_W,iAUX_B,iAUX_G,iAUX_BE,
    iBF_W,iBF_B,iBF_G,iBF_BE,
    iGATE_W,iGATE_B,
    iJF_W1,iJF_B1,iJF_G,iJF_BE,iJF_W2,iJF_B2,
    iJN_G,iJN_BE,
    iHEAD_W,iHEAD_B,
    iPQ_W,iPQ_B,
    iPA_W,iPA_B,
    iPG_W1,iPG_B1,iPG_G,iPG_BE,iPG_W2,iPG_B2
  };

  const size_t M = (size_t)M_TOT;
  uintptr_t wbase = (uintptr_t)d_ws; size_t off = 0;
  auto alloc = [&](size_t bytes)->void*{
    off = (off + 255) & ~(size_t)255;
    void* p = (void*)(wbase + off); off += bytes; return p;
  };
  // ---- bf16 buffers
  bfu* frameB  = (bfu*)alloc(M*D_*2);
  bfu* baseB   = (bfu*)alloc(M*F_*2);
  bfu* semB    = (bfu*)alloc(M*F_*2);
  bfu* bfB     = (bfu*)alloc(M*F_*2);
  bfu* vctxB   = (bfu*)alloc(M*F_*2);
  bfu* sctxB   = (bfu*)alloc(M*F_*2);
  bfu* auxembB = (bfu*)alloc(M*F_*2);
  bfu* gsemB   = (bfu*)alloc(M*F_*2);
  bfu* jointB  = (bfu*)alloc(M*F_*2);
  bfu* j1B     = (bfu*)alloc(M*512*2);
  bfu* pgeluB  = (bfu*)alloc(M*F_*2);
  bfu* qB      = (bfu*)alloc(M*D_*2);
  bfu* hl_v    = (bfu*)alloc(M*640*2);
  bfu* hl_s    = (bfu*)alloc(M*640*2);
  bfu* protoN  = (bfu*)alloc((size_t)32*D_*2);
  bfu* protoRaw= (bfu*)alloc((size_t)32*D_*2);
  bfu* sprotoN = (bfu*)alloc((size_t)32*D_*2);
  bfu* pgW2pad = (bfu*)alloc((size_t)32*F_*2);
  // ---- bf16 weights
  bfu* baseW = (bfu*)alloc((size_t)F_*D_*2);
  bfu* vWih  = (bfu*)alloc((size_t)768*256*2);
  bfu* vWhh  = (bfu*)alloc((size_t)768*256*2);
  bfu* vW1x  = (bfu*)alloc((size_t)384*256*2);
  bfu* vW1h  = (bfu*)alloc((size_t)384*256*2);
  bfu* vW2   = (bfu*)alloc((size_t)384*384*2);
  bfu* vWgs  = (bfu*)alloc((size_t)384*384*2);
  bfu* vWgl  = (bfu*)alloc((size_t)384*384*2);
  bfu* vWcs  = (bfu*)alloc((size_t)384*384*2);
  bfu* vWcl  = (bfu*)alloc((size_t)384*384*2);
  bfu* vctxW = (bfu*)alloc((size_t)256*640*2);
  bfu* sWih  = (bfu*)alloc((size_t)768*256*2);
  bfu* sWhh  = (bfu*)alloc((size_t)768*256*2);
  bfu* sW1x  = (bfu*)alloc((size_t)384*256*2);
  bfu* sW1h  = (bfu*)alloc((size_t)384*256*2);
  bfu* sW2   = (bfu*)alloc((size_t)384*384*2);
  bfu* sWgs  = (bfu*)alloc((size_t)384*384*2);
  bfu* sWgl  = (bfu*)alloc((size_t)384*384*2);
  bfu* sWcs  = (bfu*)alloc((size_t)384*384*2);
  bfu* sWcl  = (bfu*)alloc((size_t)384*384*2);
  bfu* sctxW = (bfu*)alloc((size_t)256*640*2);
  bfu* bfW   = (bfu*)alloc((size_t)256*256*2);
  bfu* gW0   = (bfu*)alloc((size_t)256*256*2);
  bfu* gW1   = (bfu*)alloc((size_t)256*256*2);
  bfu* gW2   = (bfu*)alloc((size_t)256*256*2);
  bfu* gW3   = (bfu*)alloc((size_t)256*256*2);
  bfu* jf1a  = (bfu*)alloc((size_t)512*256*2);
  bfu* jf1b  = (bfu*)alloc((size_t)512*256*2);
  bfu* jf1c  = (bfu*)alloc((size_t)512*256*2);
  bfu* jfW2  = (bfu*)alloc((size_t)256*512*2);
  bfu* headW = (bfu*)alloc((size_t)C_*256*2);
  bfu* pqW   = (bfu*)alloc((size_t)D_*256*2);
  bfu* paW   = (bfu*)alloc((size_t)D_*D_*2);
  bfu* pgW1a = (bfu*)alloc((size_t)256*256*2);
  bfu* pgW1b = (bfu*)alloc((size_t)256*256*2);
  // ---- f32 buffers
  float* norms   = (float*)alloc(M*4);
  float* dotp    = (float*)alloc(M*4);
  float* qnorm   = (float*)alloc(M*4);
  float* sims    = (float*)alloc(M*32*4);
  float* auxF    = (float*)alloc(M*A_*4);
  float* G       = (float*)alloc((size_t)(S_+E_)*F_*4);
  float* preA    = (float*)alloc(M*512*4);     // reused scratch for pre-activations
  float* bfF     = (float*)alloc(M*F_*4);
  float* sctxF   = (float*)alloc(M*F_*4);
  float* jointF  = (float*)alloc(M*F_*4);
  float* mainl   = (float*)alloc(M*C_*4);
  float* pg2     = (float*)alloc(M*32*4);
  float* qdot    = (float*)alloc(M*32*4);
  float* sprotoPre = (float*)alloc((size_t)32*D_*4);
  float* gi_v    = (float*)alloc(M*768*4);     // contiguous region also reused for qF
  float* gi_s    = (float*)alloc(M*768*4);
  float* xs_v    = (float*)alloc(M*384*4);
  float* xs_s    = (float*)alloc(M*384*4);
  float* qF      = gi_v;                        // 75MB region >= M*D_*4 (67MB); scans done by then

  auto gemm = [&](const bfu* A, const bfu* W, const float* bias, float* Out,
                  int Mm, int N, int K, int acc){
    dim3 g((unsigned)(Mm/16), (unsigned)((N + 255)/256));
    k_gemm<<<g, 256, 0, stream>>>(A, W, bias, Out, Mm, N, K, acc);
  };
  auto cvt = [&](const float* s, bfu* d, size_t n){
    k_f2bf<<<(unsigned)((n + 255)/256), 256, 0, stream>>>(s, d, n);
  };
  auto cvtsl = [&](const float* s, bfu* d, int N, int ldK, int c0, int Kd){
    k_f2bf_slice<<<(unsigned)((N*Kd + 255)/256), 256, 0, stream>>>(s, d, N, ldK, c0, Kd);
  };

  // ---- weight prep (all hot in L2 afterwards)
  cvt(PRM(iBASE_W), baseW, (size_t)F_*D_);
  cvt(PRM(iVGRU_WIH), vWih, 768*256);  cvt(PRM(iVGRU_WHH), vWhh, 768*256);
  cvt(PRM(iSGRU_WIH), sWih, 768*256);  cvt(PRM(iSGRU_WHH), sWhh, 768*256);
  cvtsl(PRM(iVSUM_W1), vW1x, 384, 512, 0, 256);  cvtsl(PRM(iVSUM_W1), vW1h, 384, 512, 256, 256);
  cvtsl(PRM(iSSUM_W1), sW1x, 384, 512, 0, 256);  cvtsl(PRM(iSSUM_W1), sW1h, 384, 512, 256, 256);
  cvt(PRM(iVSUM_W2), vW2, 384*384);  cvt(PRM(iSSUM_W2), sW2, 384*384);
  cvtsl(PRM(iVLONG_WG), vWgs, 384, 768, 0, 384);  cvtsl(PRM(iVLONG_WG), vWgl, 384, 768, 384, 384);
  cvtsl(PRM(iVLONG_WC), vWcs, 384, 768, 0, 384);  cvtsl(PRM(iVLONG_WC), vWcl, 384, 768, 384, 384);
  cvtsl(PRM(iSLONG_WG), sWgs, 384, 768, 0, 384);  cvtsl(PRM(iSLONG_WG), sWgl, 384, 768, 384, 384);
  cvtsl(PRM(iSLONG_WC), sWcs, 384, 768, 0, 384);  cvtsl(PRM(iSLONG_WC), sWcl, 384, 768, 384, 384);
  cvt(PRM(iVCTX_W), vctxW, 256*640);  cvt(PRM(iSCTX_W), sctxW, 256*640);
  cvt(PRM(iBF_W), bfW, 256*256);
  cvtsl(PRM(iGATE_W), gW0, 256, 1024, 0, 256);   cvtsl(PRM(iGATE_W), gW1, 256, 1024, 256, 256);
  cvtsl(PRM(iGATE_W), gW2, 256, 1024, 512, 256); cvtsl(PRM(iGATE_W), gW3, 256, 1024, 768, 256);
  cvtsl(PRM(iJF_W1), jf1a, 512, 768, 0, 256);    cvtsl(PRM(iJF_W1), jf1b, 512, 768, 256, 256);
  cvtsl(PRM(iJF_W1), jf1c, 512, 768, 512, 256);
  cvt(PRM(iJF_W2), jfW2, 256*512);
  cvt(PRM(iHEAD_W), headW, (size_t)C_*256);
  cvt(PRM(iPQ_W), pqW, (size_t)D_*256);
  cvt(PRM(iPA_W), paW, (size_t)D_*D_);
  cvtsl(PRM(iPG_W1), pgW1a, 256, 2*F_ + A_, 0, 256);
  cvtsl(PRM(iPG_W1), pgW1b, 256, 2*F_ + A_, 256, 256);
  k_zero_bf<<<(32*F_)/256, 256, 0, stream>>>(pgW2pad, (size_t)32*F_);
  cvt(PRM(iPG_W2), pgW2pad, (size_t)S_*F_);
  // prototype banks: normalized (for sims), raw-padded (for pa GEMM)
  k_zero_bf<<<(32*D_)/256, 256, 0, stream>>>(protoN, (size_t)32*D_);
  k_norm_rows<<<S_, 256, 0, stream>>>(stepP, protoN, S_, D_);
  k_norm_rows<<<E_, 256, 0, stream>>>(errP, protoN + (size_t)S_*D_, E_, D_);
  k_zero_bf<<<(32*D_)/256, 256, 0, stream>>>(protoRaw, (size_t)32*D_);
  cvt(stepP, protoRaw, (size_t)S_*D_);
  k_fold<<<((S_+E_)*F_ + 255)/256, 256, 0, stream>>>(stepP, errP, PRM(iSIN_W), G);

  // ---- frame prep
  k_feat_tr<<<(unsigned)((M*D_ + 255)/256), 256, 0, stream>>>(feats, frameB);
  k_rowstats<<<(unsigned)M, 256, 0, stream>>>(feats, norms, dotp);

  // ---- base projection + LN + GELU
  gemm(frameB, baseW, PRM(iBASE_B), preA, M_TOT, F_, D_, 0);
  k_ln_act<<<(unsigned)M, 256, 0, stream>>>(preA, nullptr, PRM(iBASE_G), PRM(iBASE_BE), F_, 1, nullptr, baseB);
  // ---- cosine sims vs prototype bank (one GEMM, 26 live of 32 cols)
  gemm(frameB, protoN, nullptr, sims, M_TOT, 32, D_, 0);
  k_semin<<<(unsigned)M, 256, 0, stream>>>(sims, norms, dotp, G, PRM(iSIN_W), PRM(iSIN_B), auxF, semB);

  // ---- hoist x-dependent GRU/summary projections out of the serial loop
  gemm(baseB, vWih, PRM(iVGRU_BIH), gi_v, M_TOT, 3*H_, F_, 0);
  gemm(baseB, vW1x, PRM(iVSUM_B1), xs_v, M_TOT, L_, F_, 0);
  gemm(semB,  sWih, PRM(iSGRU_BIH), gi_s, M_TOT, 3*H_, F_, 0);
  gemm(semB,  sW1x, PRM(iSSUM_B1), xs_s, M_TOT, L_, F_, 0);

  // ---- both serial scans in one kernel (grid=2, one WGP each)
  ScanArgs va{gi_v, xs_v, vWhh, PRM(iVGRU_BHH), vW1h, PRM(iVSUM_G), PRM(iVSUM_BE),
              vW2, PRM(iVSUM_B2), vWgs, vWgl, PRM(iVLONG_BG), vWcs, vWcl, PRM(iVLONG_BC), hl_v};
  ScanArgs sa{gi_s, xs_s, sWhh, PRM(iSGRU_BHH), sW1h, PRM(iSSUM_G), PRM(iSSUM_BE),
              sW2, PRM(iSSUM_B2), sWgs, sWgl, PRM(iSLONG_BG), sWcs, sWcl, PRM(iSLONG_BC), hl_s};
  k_scan<<<2, 512, 0, stream>>>(va, sa);

  // ---- context projections from packed [h|l] scan outputs
  gemm(hl_v, vctxW, PRM(iVCTX_B), preA, M_TOT, F_, 640, 0);
  k_ln_act<<<(unsigned)M, 256, 0, stream>>>(preA, nullptr, PRM(iVCTX_G), PRM(iVCTX_BE), F_, 1, nullptr, vctxB);
  gemm(hl_s, sctxW, PRM(iSCTX_B), preA, M_TOT, F_, 640, 0);
  k_ln_act<<<(unsigned)M, 256, 0, stream>>>(preA, nullptr, PRM(iSCTX_G), PRM(iSCTX_BE), F_, 1, sctxF, sctxB);
  // ---- aux embed (K=8 -> VALU), base fused
  k_lin_small<<<(unsigned)M, 256, 0, stream>>>(auxF, A_, PRM(iAUX_W), A_, 0, PRM(iAUX_B), preA, 0);
  k_ln_act<<<(unsigned)M, 256, 0, stream>>>(preA, nullptr, PRM(iAUX_G), PRM(iAUX_BE), F_, 1, nullptr, auxembB);
  gemm(baseB, bfW, PRM(iBF_B), preA, M_TOT, F_, F_, 0);
  k_ln_act<<<(unsigned)M, 256, 0, stream>>>(preA, nullptr, PRM(iBF_G), PRM(iBF_BE), F_, 1, bfF, bfB);
  // ---- fusion gate: 4 accumulating K=256 GEMMs instead of one K=1024 concat
  gemm(bfB,     gW0, PRM(iGATE_B), preA, M_TOT, F_, F_, 0);
  gemm(vctxB,   gW1, nullptr,      preA, M_TOT, F_, F_, 1);
  gemm(sctxB,   gW2, nullptr,      preA, M_TOT, F_, F_, 1);
  gemm(auxembB, gW3, nullptr,      preA, M_TOT, F_, F_, 1);
  k_gatemul<<<(unsigned)((M*F_ + 255)/256), 256, 0, stream>>>(preA, sctxF, gsemB, M*F_);
  // ---- joint fusion MLP + residual LN
  gemm(bfB,   jf1a, PRM(iJF_B1), preA, M_TOT, 512, F_, 0);
  gemm(vctxB, jf1b, nullptr,     preA, M_TOT, 512, F_, 1);
  gemm(gsemB, jf1c, nullptr,     preA, M_TOT, 512, F_, 1);
  k_ln_act<<<(unsigned)M, 256, 0, stream>>>(preA, nullptr, PRM(iJF_G), PRM(iJF_BE), 512, 1, nullptr, j1B);
  gemm(j1B, jfW2, PRM(iJF_B2), preA, M_TOT, F_, 512, 0);
  k_ln_act<<<(unsigned)M, 256, 0, stream>>>(preA, bfF, PRM(iJN_G), PRM(iJN_BE), F_, 0, jointF, jointB);
  // ---- heads
  gemm(jointB, headW, PRM(iHEAD_B), mainl, M_TOT, C_, F_, 0);
  // prototype query path
  gemm(protoRaw, paW, PRM(iPA_B), sprotoPre, 32, D_, D_, 0);
  k_zero_bf<<<(32*D_)/256, 256, 0, stream>>>(sprotoN, (size_t)32*D_);
  k_norm_rows<<<S_, 256, 0, stream>>>(sprotoPre, sprotoN, S_, D_);
  gemm(jointB, pqW, PRM(iPQ_B), qF, M_TOT, D_, F_, 0);   // qF aliases gi region (scans done)
  k_rownorm_cvt<<<(unsigned)M, 256, 0, stream>>>(qF, qB, qnorm, D_);
  gemm(qB, sprotoN, nullptr, qdot, M_TOT, 32, D_, 0);
  // prototype gate
  gemm(jointB, pgW1a, PRM(iPG_B1), preA, M_TOT, F_, F_, 0);
  gemm(sctxB,  pgW1b, nullptr,    preA, M_TOT, F_, F_, 1);
  k_lin_small<<<(unsigned)M, 256, 0, stream>>>(auxF, A_, PRM(iPG_W1), 2*F_ + A_, 2*F_, nullptr, preA, 1);
  k_ln_act<<<(unsigned)M, 256, 0, stream>>>(preA, nullptr, PRM(iPG_G), PRM(iPG_BE), F_, 1, nullptr, pgeluB);
  gemm(pgeluB, pgW2pad, nullptr, pg2, M_TOT, 32, F_, 0);
  // ---- outputs: final_logits [B,C,T] then frame_features [B,F,T]
  float* out1 = (float*)d_out;
  float* out2 = out1 + (size_t)B_*C_*T_;
  k_finalize<<<(unsigned)(((size_t)B_*C_*T_ + 255)/256), 256, 0, stream>>>(
      mainl, pg2, PRM(iPG_B2), qdot, qnorm, ids, out1);
  k_frames<<<(unsigned)(((size_t)B_*F_*T_ + 255)/256), 256, 0, stream>>>(jointF, out2);
  #undef PRM
}
